// VAE_83794811944985
// MI455X (gfx1250) — matrix operand order repaired and checked by reading, not compiled, not run on hardware
//
#include <hip/hip_runtime.h>
#include <hip/hip_bf16.h>

#define T_LEN 2048

typedef _Float16 v16h __attribute__((ext_vector_type(16)));
typedef _Float16 v8h  __attribute__((ext_vector_type(8)));
typedef float    v8f  __attribute__((ext_vector_type(8)));

__device__ __forceinline__ v8f v8f_zero() {
  v8f v = {0.f,0.f,0.f,0.f,0.f,0.f,0.f,0.f};
  return v;
}
__device__ __forceinline__ v16h cat8(v8h lo, v8h hi) {
  return __builtin_shufflevector(lo, hi, 0,1,2,3,4,5,6,7,8,9,10,11,12,13,14,15);
}

// A fragment (16x32 f16), rows m=mh*16.., K block at kcol. Row-major [rows][ldk].
// Lanes 0-15: row=lane, K kcol+0..7 / kcol+16..23 ; lanes 16-31: row=lane-16, K+8../K+24..
__device__ __forceinline__ v16h load_a(const _Float16* __restrict__ p, int ldk,
                                       int mh, int kcol, int lane) {
  int row = mh*16 + (lane & 15);
  int col = kcol + ((lane >> 4) << 3);
  const _Float16* q = p + (size_t)row*ldk + col;
  v8h lo = *(const v8h*)q;
  v8h hi = *(const v8h*)(q + 16);
  return cat8(lo, hi);
}
// B fragment (32x16): weights row-major [1536][ldk]; out cols n0..n0+15, K kcol..+31.
__device__ __forceinline__ v16h load_b(const _Float16* __restrict__ w, int ldk,
                                       int n0, int kcol, int lane) {
  int n = n0 + (lane & 15);
  int col = kcol + ((lane >> 4) << 3);
  const _Float16* q = w + (size_t)n*ldk + col;
  v8h lo = *(const v8h*)q;
  v8h hi = *(const v8h*)(q + 16);
  return cat8(lo, hi);
}
__device__ __forceinline__ v8f wmma_f16(v16h a, v16h b, v8f c) {
  return __builtin_amdgcn_wmma_f32_16x16x32_f16(false, a, false, b, (short)0, c, false, false);
}
__device__ __forceinline__ float sigmf(float x) { return 1.f/(1.f+__expf(-x)); }

// One K-block (K=32) of the gate GEMM: 3 B tiles (r/z/n), 2 M tiles.
__device__ __forceinline__ void gemm_kblock(
    const _Float16* __restrict__ aX, int ald,
    const _Float16* __restrict__ W, int ldw,
    int cw, int lane, int kcA, int kcW,
    v8f aR[2], v8f aZ[2], v8f aN[2])
{
  v16h br = load_b(W, ldw, cw,        kcW, lane);
  v16h bz = load_b(W, ldw, 512 + cw,  kcW, lane);
  v16h bn = load_b(W, ldw, 1024 + cw, kcW, lane);
#pragma unroll
  for (int mh = 0; mh < 2; ++mh) {
    v16h a = load_a(aX, ald, mh, kcA, lane);
    aR[mh] = wmma_f16(a, br, aR[mh]);
    aZ[mh] = wmma_f16(a, bz, aZ[mh]);
    aN[mh] = wmma_f16(a, bn, aN[mh]);
  }
}

// x-part: accumulates r/z and inn (input-gate n term).
template<int KXB>
__device__ __forceinline__ void gemm_x(
    const _Float16* __restrict__ aX, int ald,
    const _Float16* __restrict__ W,
    int cw, int lane, v8f aR[2], v8f aZ[2], v8f aIN[2])
{
  constexpr int LDW = KXB*32 + 512;
  if constexpr (KXB <= 8) {
#pragma unroll
    for (int kb = 0; kb < KXB; ++kb)
      gemm_kblock(aX, ald, W, LDW, cw, lane, kb*32, kb*32, aR, aZ, aIN);
  } else {
#pragma unroll 4
    for (int kb = 0; kb < KXB; ++kb)
      gemm_kblock(aX, ald, W, LDW, cw, lane, kb*32, kb*32, aR, aZ, aIN);
  }
}

// h-part: accumulates r/z and hn (hidden-gate n term). hS is always LDS.
template<int KXB>
__device__ __forceinline__ void gemm_h(
    const _Float16* __restrict__ hS,
    const _Float16* __restrict__ W,
    int cw, int lane, v8f aR[2], v8f aZ[2], v8f aHN[2])
{
  constexpr int LDW = KXB*32 + 512;
#pragma unroll 2
  for (int kb = 0; kb < 16; ++kb)
    gemm_kblock(hS, 520, W, LDW, cw, lane, kb*32, KXB*32 + kb*32, aR, aZ, aHN);
}

__device__ __forceinline__ void acc_init(v8f aR[2], v8f aZ[2], v8f aIN[2], v8f aHN[2]) {
#pragma unroll
  for (int mh = 0; mh < 2; ++mh) {
    aR[mh] = v8f_zero(); aZ[mh] = v8f_zero();
    aIN[mh] = v8f_zero(); aHN[mh] = v8f_zero();
  }
}

__device__ __forceinline__ void gru_update(
    v8f aR[2], v8f aZ[2], v8f aIN[2], v8f aHN[2],
    float bR, float bZ, float bIN, float bHN,
    float hOld[2][8], _Float16* hS, int c, int rbase,
    _Float16* outSeqT /* nullable, pre-offset by t */, int outOff)
{
#pragma unroll
  for (int mh = 0; mh < 2; ++mh) {
#pragma unroll
    for (int v = 0; v < 8; ++v) {
      float r  = sigmf(aR[mh][v] + bR);
      float z  = sigmf(aZ[mh][v] + bZ);
      float n  = tanhf(aIN[mh][v] + bIN + r*(aHN[mh][v] + bHN));
      float h2 = (1.f - z)*n + z*hOld[mh][v];
      hOld[mh][v] = h2;
      int row = mh*16 + rbase + v;
      _Float16 hh = (_Float16)h2;
      hS[row*520 + c] = hh;
      if (outSeqT) outSeqT[(size_t)row*1024 + outOff + c] = hh;
    }
  }
}

struct GruParams {
  const _Float16* X; int xld;
  const _Float16* W;
  const float* bih; const float* bhh;
  _Float16* outSeq; int outOff;     // out row stride fixed at 1024 (f0|b0 concat)
  float* hFinal;                    // stride 2048 (henc slice) or nullptr
  int reverse;
};

// One workgroup (32 waves) per GRU direction; wave w owns h columns [16w,16w+16).
template<int KXB, bool STAGE>
__global__ __launch_bounds__(1024) void gru_pair_kernel(GruParams pa, GruParams pb) {
  GruParams p = (blockIdx.x == 0) ? pa : pb;
  __shared__ __align__(16) _Float16 hS[32*520];
  __shared__ __align__(16) _Float16 xS[32*176];
  int tid = threadIdx.x, lane = tid & 31, wave = tid >> 5;
  int cw = wave << 4;
  int c = cw + (lane & 15);
  int rbase = (lane >> 4) << 3;
  float bR  = p.bih[c]        + p.bhh[c];
  float bZ  = p.bih[512 + c]  + p.bhh[512 + c];
  float bIN = p.bih[1024 + c];
  float bHN = p.bhh[1024 + c];
  for (int i = tid; i < 32*520; i += 1024) hS[i] = (_Float16)0.f;
  float hOld[2][8];
#pragma unroll
  for (int mh = 0; mh < 2; ++mh)
#pragma unroll
    for (int v = 0; v < 8; ++v) hOld[mh][v] = 0.f;
  __syncthreads();

  for (int s = 0; s < T_LEN; ++s) {
    int t = p.reverse ? (T_LEN - 1 - s) : s;
    const _Float16* Xt = p.X + (size_t)t * 32 * p.xld;
    // Opaque zero offset: loads stay loop-variant (no LICM hoist -> no scratch
    // spill of 2048-step-invariant weights) while the base pointer remains
    // traceable to a global kernarg -> global_load_b128, not flat_load.
    int wofs = 0;
    asm volatile("" : "+s"(wofs));
    const _Float16* Wt = p.W + wofs;
    v8f aR[2], aZ[2], aIN[2], aHN[2];
    acc_init(aR, aZ, aIN, aHN);
    if constexpr (STAGE) {                      // layer0: stage 32x160 panel to LDS
      for (int i = tid; i < 32*20; i += 1024) {
        int r = i / 20, c8 = (i % 20) * 8;
        *(v8h*)(xS + r*176 + c8) = *(const v8h*)(Xt + r*160 + c8);
      }
      __syncthreads();
      gemm_x<KXB>(xS, 176, Wt, cw, lane, aR, aZ, aIN);   // A from LDS
    } else {
      __syncthreads();
      gemm_x<KXB>(Xt, p.xld, Wt, cw, lane, aR, aZ, aIN); // A from global (L0)
    }
    gemm_h<KXB>(hS, Wt, cw, lane, aR, aZ, aHN);          // A from LDS
    _Float16* outT = p.outSeq ? p.outSeq + (size_t)t*32*1024 : (_Float16*)nullptr;
    gru_update(aR, aZ, aIN, aHN, bR, bZ, bIN, bHN, hOld, hS, c, rbase, outT, p.outOff);
    __syncthreads();
  }
  if (p.hFinal) {
#pragma unroll
    for (int mh = 0; mh < 2; ++mh)
#pragma unroll
      for (int v = 0; v < 8; ++v) {
        int row = mh*16 + rbase + v;
        p.hFinal[(size_t)row*2048 + c] = hOld[mh][v];
      }
  }
}

// Autoregressive decoder GRU with teacher forcing (single block).
__global__ __launch_bounds__(1024) void gru_dec_kernel(
    const _Float16* __restrict__ X, const _Float16* __restrict__ W,
    const float* __restrict__ bih, const float* __restrict__ bhh,
    const float* __restrict__ hInit,
    const float* __restrict__ fcw, const float* __restrict__ fcb,
    const unsigned char* __restrict__ tfm, const float* __restrict__ label,
    float* __restrict__ recon)
{
  __shared__ __align__(16) _Float16 hS[32*520];
  __shared__ __align__(16) _Float16 xS[32*176];
  __shared__ float prevS[32*4];
  __shared__ float partS[1024];
  int tid = threadIdx.x, lane = tid & 31, wave = tid >> 5;
  int cw = wave << 4;
  int c = cw + (lane & 15);
  int rbase = (lane >> 4) << 3;
  float bR  = bih[c]        + bhh[c];
  float bZ  = bih[512 + c]  + bhh[512 + c];
  float bIN = bih[1024 + c];
  float bHN = bhh[1024 + c];
  for (int i = tid; i < 32*520; i += 1024) {
    int r = i / 520, cc = i % 520;
    hS[i] = (cc < 512) ? (_Float16)hInit[r*512 + cc] : (_Float16)0.f;
  }
  float hOld[2][8];
#pragma unroll
  for (int mh = 0; mh < 2; ++mh)
#pragma unroll
    for (int v = 0; v < 8; ++v) {
      int row = mh*16 + rbase + v;
      hOld[mh][v] = hInit[row*512 + c];
    }
  if (tid < 128) prevS[tid] = 0.5f;
  __syncthreads();

  for (int t = 0; t < T_LEN; ++t) {
    const _Float16* Xt = X + (size_t)t * 32 * 160;
    int wofs = 0;
    asm volatile("" : "+s"(wofs));   // defeat cross-step hoist, keep global AS
    const _Float16* Wt = W + wofs;
    for (int i = tid; i < 32*160; i += 1024) {   // stage + patch prev (cols 128..131)
      int b = i / 160, cc = i % 160;
      _Float16 v = (cc >= 128 && cc < 132) ? (_Float16)prevS[b*4 + (cc - 128)] : Xt[i];
      xS[b*176 + cc] = v;
    }
    __syncthreads();

    v8f aR[2], aZ[2], aIN[2], aHN[2];
    acc_init(aR, aZ, aIN, aHN);
    gemm_x<5>(xS, 176, Wt, cw, lane, aR, aZ, aIN);
    gemm_h<5>(hS, Wt, cw, lane, aR, aZ, aHN);
    gru_update(aR, aZ, aIN, aHN, bR, bZ, bIN, bHN, hOld, hS, c, rbase, nullptr, 0);
    __syncthreads();

    { // out = h2 @ fc_dec_w^T + b : 1024-way partial dots
      int b = tid >> 5, ic = (tid >> 3) & 3, seg = tid & 7;
      const float* wrow = fcw + ic*512;
      int k0 = seg * 64;
      float acc = 0.f;
      for (int k = 0; k < 64; ++k) acc += (float)hS[b*520 + k0 + k] * wrow[k0 + k];
      partS[tid] = acc;
    }
    __syncthreads();
    if (tid < 128) {
      int b = tid >> 2, ic = tid & 3;
      float o = fcb[ic];
      for (int s2 = 0; s2 < 8; ++s2) o += partS[(b << 5) + (ic << 3) + s2];
      recon[(size_t)b*T_LEN*4 + (size_t)t*4 + ic] = o;
      prevS[b*4 + ic] = tfm[t] ? label[(size_t)b*T_LEN*4 + (size_t)t*4 + ic] : o;
    }
    __syncthreads();
  }
}

// ---- prep kernels ----
__global__ void prep_w(const float* __restrict__ wih, const float* __restrict__ whh,
                       _Float16* __restrict__ dst, int Ik, int kxpad, int ldw) {
  int idx = blockIdx.x*blockDim.x + threadIdx.x;
  if (idx >= 1536*ldw) return;
  int row = idx / ldw, k = idx % ldw;
  float v;
  if (k < Ik) v = wih[(size_t)row*Ik + k];
  else if (k < kxpad) v = 0.f;
  else v = whh[(size_t)row*512 + (k - kxpad)];
  dst[idx] = (_Float16)v;
}

__global__ void prep_x(const float* __restrict__ cond, const float* __restrict__ label,
                       const float* __restrict__ pos,
                       _Float16* __restrict__ xenc, _Float16* __restrict__ xdec) {
  int idx = blockIdx.x*blockDim.x + threadIdx.x;
  if (idx >= T_LEN*32*160) return;
  int t = idx / (32*160), rem = idx % (32*160);
  int b = rem / 160, c = rem % 160;
  float ve = 0.f, vd = 0.f;
  if (c < 128) { float v = cond[(size_t)b*T_LEN*128 + (size_t)t*128 + c]; ve = vd = v; }
  else if (c < 132) { ve = label[(size_t)b*T_LEN*4 + (size_t)t*4 + (c - 128)]; vd = 0.f; }
  else if (c < 136) { float v = pos[(t & 7)*4 + (c - 132)]; ve = vd = v; }
  xenc[idx] = (_Float16)ve;
  xdec[idx] = (_Float16)vd;
}

// ---- latent path (tiny, precision sensitive -> f32 VALU, fully parallel) ----
__global__ void latent1(const float* __restrict__ henc, const float* __restrict__ w,
                        const float* __restrict__ b, float* __restrict__ hmid) {
  int idx = blockIdx.x*blockDim.x + threadIdx.x;
  if (idx >= 32*512) return;
  int bb = idx >> 9, j = idx & 511;
  const float* x = henc + (size_t)bb*2048;
  const float* wr = w + (size_t)j*2048;
  float acc = b[j];
  for (int k = 0; k < 2048; ++k) acc += x[k]*wr[k];
  hmid[idx] = fmaxf(acc, 0.f);
}
__global__ void latent2(const float* __restrict__ hmid,
                        const float* __restrict__ wmu, const float* __restrict__ bmu,
                        const float* __restrict__ wls, const float* __restrict__ bls,
                        const float* __restrict__ eps,
                        float* __restrict__ outMu, float* __restrict__ outLs,
                        float* __restrict__ z) {
  int idx = blockIdx.x*blockDim.x + threadIdx.x;
  if (idx >= 32*512) return;
  int bb = idx >> 9, j = idx & 511;
  const float* x = hmid + (size_t)bb*512;
  const float* wm = wmu + (size_t)j*512;
  const float* wl = wls + (size_t)j*512;
  float mu = bmu[j], ls = bls[j];
  for (int k = 0; k < 512; ++k) { mu += x[k]*wm[k]; ls += x[k]*wl[k]; }
  outMu[idx] = mu; outLs[idx] = ls;
  z[idx] = mu + eps[idx]*__expf(ls);
}
__global__ void latent3(const float* __restrict__ z, const float* __restrict__ w,
                        const float* __restrict__ b, float* __restrict__ hdec) {
  int idx = blockIdx.x*blockDim.x + threadIdx.x;
  if (idx >= 32*512) return;
  int bb = idx >> 9, j = idx & 511;
  const float* x = z + (size_t)bb*512;
  const float* wr = w + (size_t)j*512;
  float acc = b[j];
  for (int k = 0; k < 512; ++k) acc += x[k]*wr[k];
  hdec[idx] = acc;
}

extern "C" void kernel_launch(void* const* d_in, const int* in_sizes, int n_in,
                              void* d_out, int out_size, void* d_ws, size_t ws_size,
                              hipStream_t stream) {
  (void)in_sizes; (void)n_in; (void)out_size; (void)ws_size;
  const float* cond      = (const float*)d_in[0];
  const float* label     = (const float*)d_in[1];
  const float* eps       = (const float*)d_in[2];
  const unsigned char* tfm = (const unsigned char*)d_in[3]; // bool mask (1 byte)
  const float* pos       = (const float*)d_in[4];
  const float* wih0f = (const float*)d_in[5],  *whh0f = (const float*)d_in[6];
  const float* bih0f = (const float*)d_in[7],  *bhh0f = (const float*)d_in[8];
  const float* wih0b = (const float*)d_in[9],  *whh0b = (const float*)d_in[10];
  const float* bih0b = (const float*)d_in[11], *bhh0b = (const float*)d_in[12];
  const float* wih1f = (const float*)d_in[13], *whh1f = (const float*)d_in[14];
  const float* bih1f = (const float*)d_in[15], *bhh1f = (const float*)d_in[16];
  const float* wih1b = (const float*)d_in[17], *whh1b = (const float*)d_in[18];
  const float* bih1b = (const float*)d_in[19], *bhh1b = (const float*)d_in[20];
  const float* shrink_w = (const float*)d_in[21], *shrink_b = (const float*)d_in[22];
  const float* fc_mu_w  = (const float*)d_in[23], *fc_mu_b  = (const float*)d_in[24];
  const float* fc_ls_w  = (const float*)d_in[25], *fc_ls_b  = (const float*)d_in[26];
  const float* fc_z_w   = (const float*)d_in[27], *fc_z_b   = (const float*)d_in[28];
  const float* dec_wih  = (const float*)d_in[29], *dec_whh  = (const float*)d_in[30];
  const float* dec_bih  = (const float*)d_in[31], *dec_bhh  = (const float*)d_in[32];
  const float* fc_dec_w = (const float*)d_in[33], *fc_dec_b = (const float*)d_in[34];
  float* outF = (float*)d_out;
  float* outMu = outF + 32*T_LEN*4;           // recon is [32,2048,4]
  float* outLs = outMu + 32*512;

  char* ws = (char*)d_ws;
  size_t off = 0;
  auto take = [&](size_t bytes) { void* p = ws + off; off += (bytes + 255) & ~(size_t)255; return p; };
  _Float16* W0f  = (_Float16*)take((size_t)1536*672*2);
  _Float16* W0b  = (_Float16*)take((size_t)1536*672*2);
  _Float16* W1f  = (_Float16*)take((size_t)1536*1536*2);
  _Float16* W1b  = (_Float16*)take((size_t)1536*1536*2);
  _Float16* Wd   = (_Float16*)take((size_t)1536*672*2);
  _Float16* Xenc = (_Float16*)take((size_t)T_LEN*32*160*2);
  _Float16* Xdec = (_Float16*)take((size_t)T_LEN*32*160*2);
  _Float16* X1   = (_Float16*)take((size_t)T_LEN*32*1024*2);  // f0|b0 sequences (layer1 input)
  float* henc = (float*)take((size_t)32*2048*4);
  float* hmid = (float*)take((size_t)32*512*4);
  float* zbuf = (float*)take((size_t)32*512*4);
  float* hdec = (float*)take((size_t)32*512*4);

  // ---- prep: pack weights to f16 (K = [x_pad | h]) and inputs to f16 panels ----
  int nw0 = 1536*672, nw1 = 1536*1536;
  prep_w<<<(nw0+255)/256, 256, 0, stream>>>(wih0f, whh0f, W0f, 136, 160, 672);
  prep_w<<<(nw0+255)/256, 256, 0, stream>>>(wih0b, whh0b, W0b, 136, 160, 672);
  prep_w<<<(nw1+255)/256, 256, 0, stream>>>(wih1f, whh1f, W1f, 1024, 1024, 1536);
  prep_w<<<(nw1+255)/256, 256, 0, stream>>>(wih1b, whh1b, W1b, 1024, 1024, 1536);
  prep_w<<<(nw0+255)/256, 256, 0, stream>>>(dec_wih, dec_whh, Wd, 136, 160, 672);
  prep_x<<<(T_LEN*32*160+255)/256, 256, 0, stream>>>(cond, label, pos, Xenc, Xdec);

  // ---- encoder layer 0: forward + backward concurrently ----
  GruParams p0f; p0f.X = Xenc; p0f.xld = 160;
  p0f.W = W0f; p0f.bih = bih0f; p0f.bhh = bhh0f;
  p0f.outSeq = X1; p0f.outOff = 0; p0f.hFinal = henc; p0f.reverse = 0;
  GruParams p0b = p0f; p0b.W = W0b; p0b.bih = bih0b; p0b.bhh = bhh0b;
  p0b.outOff = 512; p0b.hFinal = henc + 512; p0b.reverse = 1;
  gru_pair_kernel<5, true><<<2, 1024, 0, stream>>>(p0f, p0b);

  // ---- encoder layer 1 ----
  GruParams p1f; p1f.X = X1; p1f.xld = 1024;
  p1f.W = W1f; p1f.bih = bih1f; p1f.bhh = bhh1f;
  p1f.outSeq = nullptr; p1f.outOff = 0; p1f.hFinal = henc + 1024; p1f.reverse = 0;
  GruParams p1b = p1f; p1b.W = W1b; p1b.bih = bih1b; p1b.bhh = bhh1b;
  p1b.hFinal = henc + 1536; p1b.reverse = 1;
  gru_pair_kernel<32, false><<<2, 1024, 0, stream>>>(p1f, p1b);

  // ---- latent: hmid -> (mu, log_std) -> z -> hdec ----
  latent1<<<64, 256, 0, stream>>>(henc, shrink_w, shrink_b, hmid);
  latent2<<<64, 256, 0, stream>>>(hmid, fc_mu_w, fc_mu_b, fc_ls_w, fc_ls_b, eps,
                                  outMu, outLs, zbuf);
  latent3<<<64, 256, 0, stream>>>(zbuf, fc_z_w, fc_z_b, hdec);

  // ---- decoder ----
  gru_dec_kernel<<<1, 1024, 0, stream>>>(Xdec, Wd, dec_bih, dec_bhh, hdec,
                                         fc_dec_w, fc_dec_b, tfm, label, outF);
}
